// Encoder_81346680586940
// MI455X (gfx1250) — compile-verified, matrix operand order checked by reading, not run on hardware
//
#include <hip/hip_runtime.h>
#include <math.h>

// ---------------------------------------------------------------------------
// CfC (closed-form continuous-time) RNN, NCP-wired, for MI455X (gfx1250).
// Strategy: 16 workgroups x 16 batch rows; per-timestep GEMMs via
// V_WMMA_F32_16X16X32_F16 on pre-packed f16 weight fragments; all recurrent
// state in LDS; gate matmuls folded (Wa+Wb); final FC also via WMMA.
// ---------------------------------------------------------------------------

typedef _Float16 half8 __attribute__((ext_vector_type(8)));
typedef _Float16 v16h  __attribute__((ext_vector_type(16)));
typedef float    v8f   __attribute__((ext_vector_type(8)));

union AFrag { v16h v; half8 h[2]; };

#define SEQT 1024

// Layer 0: fin=64, hid=135, K=199 -> 7 k-tiles (224), 9 n-tiles (144)
#define K0 199
#define H0 135
#define KT0 7
#define NT0 9
#define KS0 232      // padded row stride (halfs), 464B = 16B-aligned rows
#define SELF0 64     // column where own h starts in X0
// Layer 1: fin=135, hid=89, K=224 -> 7 k-tiles, 6 n-tiles (96)
#define H1 89
#define KT1 7
#define NT1 6
#define KS1 232
#define SELF1 135
// Layer 2: fin=89, hid=32, K=121 -> 4 k-tiles (128), 2 n-tiles
#define K2 121
#define H2 32
#define KT2 4
#define NT2 2
#define KS2 136
#define SELF2 89
// Final FC: K=256 -> 8 k-tiles, N=256 -> 16 n-tiles
#define KTF 8
#define NTF 16
#define KSF 264

#define NPAD 144     // U (pre-activation) column stride; max padded hid

// LDS carve (bytes); total 63232 < 64KB static limit
#define OFF_X0  0        // 16*232*2 = 7424
#define OFF_X1  7424     // 7424
#define OFF_X2  14848    // 16*136*2 = 4352
#define OFF_H32 19200    // 16*256*4 = 16384  (f32 state, already concat order)
#define OFF_U   35584    // 3*16*144*4 = 27648 ; Xf (16*264*2=8448) aliases this
#define SMEM_BYTES 63232

// packed fragment sizes (halfs)
#define PK0_N (3*KT0*NT0*512)   // 96768
#define PK1_N (3*KT1*NT1*512)   // 64512
#define PK2_N (3*KT2*NT2*512)   // 12288
#define PKF_N (1*KTF*NTF*512)   // 65536

// ---------------------------------------------------------------------------
// Pack kernel: W[hid, K] (f32, optional mask, optional Wa+Wb fold) ->
// per-(mat,ktile,ntile) 32-lane x 16-half B fragments matching the
// V_WMMA_F32_16X16X32_F16 B layout:
//   lane L: n = nt*16 + (L&15);  element i: k = kt*32 + (L>>4)*8 + (i&7) + (i>=8 ? 16 : 0)
// ---------------------------------------------------------------------------
__global__ void pack_kernel(const float* __restrict__ W1,
                            const float* __restrict__ W2,
                            const float* __restrict__ Wa,
                            const float* __restrict__ Wb,
                            const unsigned char* __restrict__ mask,
                            _Float16* __restrict__ dst,
                            int hid, int K, int KT, int NT, int nMats)
{
    const int total = nMats * KT * NT * 512;
    for (int tid = blockIdx.x * blockDim.x + threadIdx.x; tid < total;
         tid += gridDim.x * blockDim.x) {
        int i    = tid & 15;
        int lane = (tid >> 4) & 31;
        int rest = tid >> 9;
        int nt   = rest % NT;
        int kt   = (rest / NT) % KT;
        int mat  = rest / (NT * KT);
        int k = kt * 32 + ((lane >> 4) << 3) + (i & 7) + ((i >= 8) ? 16 : 0);
        int n = nt * 16 + (lane & 15);
        float v = 0.0f;
        if (k < K && n < hid) {
            int idx = n * K + k;
            if (mat == 0)      v = W1[idx] * (mask ? (mask[idx] ? 1.0f : 0.0f) : 1.0f);
            else if (mat == 1) v = W2[idx] * (mask[idx] ? 1.0f : 0.0f);
            else               v = Wa[idx] + Wb[idx];   // folded gate weights
        }
        dst[tid] = (_Float16)v;
    }
}

// ---------------------------------------------------------------------------
// One layer's 3 GEMMs: U[mat][16][hid_pad] = X[16][K] * Wmat^T, via WMMA.
// 8 waves strip-mine the NMATS*NT (mat, n-tile) pairs.
// ---------------------------------------------------------------------------
template<int KT, int NT, int NMATS, int KS>
__device__ __forceinline__ void tile_matmul(const _Float16* __restrict__ Xs,
                                            const _Float16* __restrict__ pk,
                                            float* __restrict__ U)
{
    const int lane = threadIdx.x & 31;
    const int wid  = threadIdx.x >> 5;
    const int lm   = lane & 15;   // A: M row ; C/D: N col
    const int hi   = lane >> 4;

    // Preload A fragments (row lm of X, split-K halves per ISA layout)
    AFrag a[KT];
    const _Float16* xrow = Xs + lm * KS + hi * 8;
#pragma unroll
    for (int kt = 0; kt < KT; ++kt) {
        a[kt].h[0] = *(const half8*)(xrow + kt * 32);
        a[kt].h[1] = *(const half8*)(xrow + kt * 32 + 16);
    }

    for (int p = wid; p < NMATS * NT; p += 8) {
        const int mat = p / NT;
        const int nt  = p % NT;
        v8f acc = {0.f, 0.f, 0.f, 0.f, 0.f, 0.f, 0.f, 0.f};
        const _Float16* bp = pk + ((size_t)(mat * KT) * NT + nt) * 512 + lane * 16;
#pragma unroll
        for (int kt = 0; kt < KT; ++kt) {
            AFrag b;
            b.h[0] = *(const half8*)(bp);
            b.h[1] = *(const half8*)(bp + 8);
            bp += (size_t)NT * 512;
            acc = __builtin_amdgcn_wmma_f32_16x16x32_f16(
                false, a[kt].v, false, b.v, (short)0, acc, false, false);
        }
        // D layout: VGPR r -> row hi*8+r, lane&15 -> col
        float* urow = U + (mat * 16 + hi * 8) * NPAD + nt * 16 + lm;
#pragma unroll
        for (int r = 0; r < 8; ++r) urow[r * NPAD] = acc[r];
    }
}

// ---------------------------------------------------------------------------
// CfC cell elementwise: h = tanh(u1) + sig(ug)*(tanh(u2)-tanh(u1))
// Writes new h into own X (f16), next layer's X (f16), and f32 state.
// ---------------------------------------------------------------------------
__device__ __forceinline__ void cell_update(int hid,
    const float* __restrict__ b1, const float* __restrict__ b2,
    const float* __restrict__ ba, const float* __restrict__ bb,
    const float* __restrict__ U,
    _Float16* Xself, int KSself, int selfOff,
    _Float16* Xnext, int KSnext,
    float* h32, int cOff)
{
    for (int j = threadIdx.x; j < 16 * hid; j += 256) {
        int m = j / hid;
        int n = j - m * hid;
        float u1 = U[m * NPAD + n]        + b1[n];
        float u2 = U[(16 + m) * NPAD + n] + b2[n];
        float ug = U[(32 + m) * NPAD + n] + ba[n] + bb[n];
        float f1 = tanhf(u1);
        float f2 = tanhf(u2);
        float tg = 1.0f / (1.0f + __expf(-ug));
        float h  = f1 + tg * (f2 - f1);
        Xself[m * KSself + selfOff + n] = (_Float16)h;
        if (Xnext) Xnext[m * KSnext + n] = (_Float16)h;
        h32[m * 256 + cOff + n] = h;
    }
}

// ---------------------------------------------------------------------------
// Main persistent recurrent kernel: one workgroup owns 16 batch rows for all
// 1024 timesteps, then applies the final FC via WMMA straight to d_out.
// ---------------------------------------------------------------------------
__global__ __launch_bounds__(256) void cfc_forward(
    const float* __restrict__ x,
    const _Float16* __restrict__ pk0,
    const _Float16* __restrict__ pk1,
    const _Float16* __restrict__ pk2,
    const _Float16* __restrict__ pkfc,
    const float* __restrict__ b1_0, const float* __restrict__ b2_0,
    const float* __restrict__ ba_0, const float* __restrict__ bb_0,
    const float* __restrict__ b1_1, const float* __restrict__ b2_1,
    const float* __restrict__ ba_1, const float* __restrict__ bb_1,
    const float* __restrict__ b1_2, const float* __restrict__ b2_2,
    const float* __restrict__ ba_2, const float* __restrict__ bb_2,
    const float* __restrict__ fc_b,
    float* __restrict__ out)
{
    __shared__ __align__(16) unsigned char smem[SMEM_BYTES];
    _Float16* X0  = (_Float16*)(smem + OFF_X0);
    _Float16* X1  = (_Float16*)(smem + OFF_X1);
    _Float16* X2  = (_Float16*)(smem + OFF_X2);
    float*    h32 = (float*)   (smem + OFF_H32);
    float*    U   = (float*)   (smem + OFF_U);
    _Float16* Xf  = (_Float16*)(smem + OFF_U);   // alias: only used after loop

    const int tid = threadIdx.x;
    const int wg  = blockIdx.x;

    // Zero X buffers (h = 0 init; padding columns stay zero forever) + state.
    for (int j = tid; j < 16 * KS0; j += 256) X0[j] = (_Float16)0.0f;
    for (int j = tid; j < 16 * KS1; j += 256) X1[j] = (_Float16)0.0f;
    for (int j = tid; j < 16 * KS2; j += 256) X2[j] = (_Float16)0.0f;
    for (int j = tid; j < 16 * 256; j += 256) h32[j] = 0.0f;
    __syncthreads();

    const float* xbase = x + (size_t)wg * 16 * SEQT * 64;

    for (int t = 0; t < SEQT; ++t) {
        // Stage x_t into X0[:, 0:64) as f16 (cols 64:199 hold h0 from t-1)
        for (int j = tid; j < 16 * 64; j += 256) {
            int m = j >> 6, c = j & 63;
            X0[m * KS0 + c] = (_Float16)xbase[((size_t)m * SEQT + t) * 64 + c];
        }
        __syncthreads();

        tile_matmul<KT0, NT0, 3, KS0>(X0, pk0, U);
        __syncthreads();
        cell_update(H0, b1_0, b2_0, ba_0, bb_0, U, X0, KS0, SELF0, X1, KS1, h32, 0);
        __syncthreads();

        tile_matmul<KT1, NT1, 3, KS1>(X1, pk1, U);
        __syncthreads();
        cell_update(H1, b1_1, b2_1, ba_1, bb_1, U, X1, KS1, SELF1, X2, KS2, h32, 135);
        __syncthreads();

        tile_matmul<KT2, NT2, 3, KS2>(X2, pk2, U);
        __syncthreads();
        cell_update(H2, b1_2, b2_2, ba_2, bb_2, U, X2, KS2, SELF2, nullptr, 0, h32, 224);
        __syncthreads();
    }

    // Final FC: out[16,256] = hn[16,256] @ fc_W^T + fc_b  (hn == h32 layout)
    for (int j = tid; j < 16 * 256; j += 256) {
        int m = j >> 8, c = j & 255;
        Xf[m * KSF + c] = (_Float16)h32[m * 256 + c];
    }
    __syncthreads();
    {
        const int lane = tid & 31;
        const int wid  = tid >> 5;
        const int lm   = lane & 15;
        const int hi   = lane >> 4;
        AFrag a[KTF];
        const _Float16* xrow = Xf + lm * KSF + hi * 8;
#pragma unroll
        for (int kt = 0; kt < KTF; ++kt) {
            a[kt].h[0] = *(const half8*)(xrow + kt * 32);
            a[kt].h[1] = *(const half8*)(xrow + kt * 32 + 16);
        }
        for (int nt = wid; nt < NTF; nt += 8) {
            v8f acc = {0.f, 0.f, 0.f, 0.f, 0.f, 0.f, 0.f, 0.f};
            const _Float16* bp = pkfc + (size_t)nt * 512 + lane * 16;
#pragma unroll
            for (int kt = 0; kt < KTF; ++kt) {
                AFrag b;
                b.h[0] = *(const half8*)(bp);
                b.h[1] = *(const half8*)(bp + 8);
                bp += (size_t)NTF * 512;
                acc = __builtin_amdgcn_wmma_f32_16x16x32_f16(
                    false, a[kt].v, false, b.v, (short)0, acc, false, false);
            }
#pragma unroll
            for (int r = 0; r < 8; ++r) {
                int row = hi * 8 + r;
                int col = nt * 16 + lm;
                out[((size_t)(wg * 16 + row)) * 256 + col] = acc[r] + fc_b[col];
            }
        }
    }
}

// ---------------------------------------------------------------------------
// Host entry
// ---------------------------------------------------------------------------
extern "C" void kernel_launch(void* const* d_in, const int* in_sizes, int n_in,
                              void* d_out, int out_size, void* d_ws, size_t ws_size,
                              hipStream_t stream)
{
    (void)in_sizes; (void)n_in; (void)out_size; (void)ws_size;

    // setup_inputs() dict order: x, then per layer l: mask, W1, W2, Wa, Wb,
    // b1, b2, ba, bb  (9 entries), then fc_W, fc_b.
    const float* x = (const float*)d_in[0];
    auto Mk = [&](int l) { return (const unsigned char*)d_in[1 + 9 * l + 0]; };
    auto Wp = [&](int l, int k) { return (const float*)d_in[1 + 9 * l + 1 + k]; }; // k=0..3: W1,W2,Wa,Wb
    auto Bp = [&](int l, int k) { return (const float*)d_in[1 + 9 * l + 5 + k]; }; // k=0..3: b1,b2,ba,bb
    const float* fcW = (const float*)d_in[28];
    const float* fcb = (const float*)d_in[29];

    _Float16* pk0  = (_Float16*)d_ws;
    _Float16* pk1  = pk0 + PK0_N;
    _Float16* pk2  = pk1 + PK1_N;
    _Float16* pkfc = pk2 + PK2_N;

    pack_kernel<<<(PK0_N + 255) / 256, 256, 0, stream>>>(
        Wp(0,0), Wp(0,1), Wp(0,2), Wp(0,3), Mk(0), pk0, H0, K0, KT0, NT0, 3);
    pack_kernel<<<(PK1_N + 255) / 256, 256, 0, stream>>>(
        Wp(1,0), Wp(1,1), Wp(1,2), Wp(1,3), Mk(1), pk1, H1, 224, KT1, NT1, 3);
    pack_kernel<<<(PK2_N + 255) / 256, 256, 0, stream>>>(
        Wp(2,0), Wp(2,1), Wp(2,2), Wp(2,3), Mk(2), pk2, H2, K2, KT2, NT2, 3);
    pack_kernel<<<(PKF_N + 255) / 256, 256, 0, stream>>>(
        fcW, nullptr, nullptr, nullptr, nullptr, pkfc, 256, 256, KTF, NTF, 1);

    cfc_forward<<<16, 256, 0, stream>>>(
        x, pk0, pk1, pk2, pkfc,
        Bp(0,0), Bp(0,1), Bp(0,2), Bp(0,3),
        Bp(1,0), Bp(1,1), Bp(1,2), Bp(1,3),
        Bp(2,0), Bp(2,1), Bp(2,2), Bp(2,3),
        fcb, (float*)d_out);
}
